// WindowTransformerBlock_206158430372
// MI455X (gfx1250) — matrix operand order repaired and checked by reading, not compiled, not run on hardware
//
#include <hip/hip_runtime.h>
#include <math.h>

// ---------------------------------------------------------------------------
// SwinV2 window-attention block for MI455X (gfx1250), wave32 + WMMA f16.
// ~840 GFLOP total, compute-bound (min HBM traffic ~300MB ~= 13us @ 23.3TB/s),
// so all heavy math runs through v_wmma_f32_16x16x32_f16 with f32 accum, and
// the GEMM K-loop streams tiles with GLOBAL_LOAD_ASYNC_TO_LDS_B128 (ASYNCcnt)
// double-buffered so the TDM-style fill overlaps the matrix pipes.
// ---------------------------------------------------------------------------

typedef _Float16 h16;
typedef __attribute__((ext_vector_type(16))) _Float16 v16h;
typedef __attribute__((ext_vector_type(8)))  _Float16 v8h;
typedef __attribute__((ext_vector_type(8)))  float    v8f;
typedef __attribute__((ext_vector_type(4)))  float    v4f;

#define TOK     32768      // B*TBINS*SPATIAL*SPATIAL tokens
#define DIMC    1024
#define HEADS_  16
#define NWIN    512        // windows total
#define HIDDEN_ 4096

// ---------------- WMMA fragment helpers (gfx1250 wave32 layouts) -----------
// A 16x32 f16: lane holds row m=lane&15; halves e<8 at K = k0+(lane>>4)*8+e,
//              halves e>=8 at K = k0+16+(lane>>4)*8+(e-8).
__device__ __forceinline__ v16h load_a_frag(const h16* base, int ldh, int m0, int k0) {
  int lane = threadIdx.x & 31;
  const h16* p = base + (size_t)(m0 + (lane & 15)) * ldh + k0 + ((lane >> 4) << 3);
  v8h lo = *(const v8h*)(p);
  v8h hi = *(const v8h*)(p + 16);
  v16h r;
#pragma unroll
  for (int i = 0; i < 8; ++i) { r[i] = lo[i]; r[i + 8] = hi[i]; }
  return r;
}

// B 32x16 f16 (KxN): lane holds col n=n0+(lane&15); halves e<8 at K = k0+(lane>>4)*8+e,
//                    halves e>=8 at K = k0+16+(lane>>4)*8+(e-8).
__device__ __forceinline__ v16h load_b_frag(const h16* base, int ldh, int n0, int k0) {
  int lane = threadIdx.x & 31;
  const h16* p = base + (size_t)(n0 + (lane & 15)) * ldh + k0 + ((lane >> 4) << 3);
  v8h lo = *(const v8h*)(p);
  v8h hi = *(const v8h*)(p + 16);
  v16h r;
#pragma unroll
  for (int i = 0; i < 8; ++i) { r[i] = lo[i]; r[i + 8] = hi[i]; }
  return r;
}

__device__ __forceinline__ v8f wmma16(v16h a, v16h b, v8f c) {
  return __builtin_amdgcn_wmma_f32_16x16x32_f16(false, a, false, b, (short)0, c, false, false);
}

// C/D f32 16x16: lane holds n = lane&15, rows m = (lane>>4)*8 + v, v=0..7.

// -------------- CDNA5 async global->LDS path (inline asm, ASYNCcnt) --------
__device__ __forceinline__ unsigned lds_addr_of(const void* p) {
  return (unsigned)(uintptr_t)(const __attribute__((address_space(3))) void*)p;
}

// per-lane 16B: LDS[vdst] = MEM[vaddr]; tracked by ASYNCcnt (in-order loads)
__device__ __forceinline__ void async_copy_b128(const h16* g, const h16* l) {
  asm volatile("global_load_async_to_lds_b128 %0, %1, off"
               :: "v"(lds_addr_of(l)), "v"(g)
               : "memory");
}

template <int N>
__device__ __forceinline__ void wait_asynccnt() {
  asm volatile("s_wait_asynccnt %0" :: "n"(N) : "memory");
}

// ------------------------------ small kernels ------------------------------
__global__ void k_f32_to_f16(const float* __restrict__ src, h16* __restrict__ dst, int n4) {
  int i = blockIdx.x * blockDim.x + threadIdx.x;
  if (i < n4) {
    v4f v = *(const v4f*)(src + (size_t)i * 4);
    h16* d = dst + (size_t)i * 4;
    d[0] = (h16)v.x; d[1] = (h16)v.y; d[2] = (h16)v.z; d[3] = (h16)v.w;
  }
}

__global__ void k_build_qkv_bias(const float* __restrict__ qb, const float* __restrict__ vb,
                                 float* __restrict__ out) {
  int n = blockIdx.x * blockDim.x + threadIdx.x;
  if (n < 3072) {
    float v = 0.f;
    if (n < 1024) v = qb[n];
    else if (n >= 2048) v = vb[n - 2048];
    out[n] = v;
  }
}

// CPB MLP: tbl[t][h] = 16*sigmoid( relu(c(t)@w1^T + b1) @ w2^T ), t in [0,225)
__global__ void k_cpb(const float* __restrict__ w1, const float* __restrict__ b1,
                      const float* __restrict__ w2, float* __restrict__ tbl) {
  __shared__ float hid[512];
  int t = blockIdx.x;                 // 0..224
  int i = t / 15, j = t % 15;
  float ah = (i - 7) / 7.0f, aw = (j - 7) / 7.0f;
  float sh = (ah > 0.f) ? 1.f : ((ah < 0.f) ? -1.f : 0.f);
  float sw = (aw > 0.f) ? 1.f : ((aw < 0.f) ? -1.f : 0.f);
  float ch = sh * log2f(1.f + fabsf(ah));
  float cw = sw * log2f(1.f + fabsf(aw));
  for (int k = threadIdx.x; k < 512; k += blockDim.x)
    hid[k] = fmaxf(0.f, ch * w1[k * 2] + cw * w1[k * 2 + 1] + b1[k]);
  __syncthreads();
  if (threadIdx.x < 16) {
    int h = threadIdx.x;
    float s = 0.f;
    for (int k = 0; k < 512; ++k) s += hid[k] * w2[h * 512 + k];
    tbl[t * 16 + h] = 16.f / (1.f + expf(-s));
  }
}

// cyclic shift + window partition fused into the f32->f16 gather
__global__ void k_gather_x(const float* __restrict__ x, h16* __restrict__ xw) {
  int r = blockIdx.x;                 // window-order row 0..32767
  int win = r >> 6, t = r & 63;
  int bimg = win >> 2, wimg = win & 3;
  int wy = wimg >> 1, wx = wimg & 1;
  int iy = t >> 3, ix = t & 7;
  int y  = (wy * 8 + iy + 4) & 15;
  int xc = (wx * 8 + ix + 4) & 15;
  const float* src = x + ((size_t)bimg * 256 + y * 16 + xc) * DIMC;
  h16* dst = xw + (size_t)r * DIMC;
  for (int c = threadIdx.x * 4; c < DIMC; c += blockDim.x * 4) {
    v4f v = *(const v4f*)(src + c);
    dst[c] = (h16)v.x; dst[c + 1] = (h16)v.y; dst[c + 2] = (h16)v.z; dst[c + 3] = (h16)v.w;
  }
}

// ------------------------------ WMMA GEMM ----------------------------------
// C[M,N] = A[M,K](f16) @ W[N,K]^T(f16) + bias ; EPI 0: f16 out, 1: f32 out,
// 2: exact GELU then f16 out. 256 threads = 8 waves (2x4), tile 128x128x32.
// Double-buffered LDS fed by global_load_async_to_lds_b128: stage s+1 streams
// in (4 async ops/thread, ASYNCcnt) while stage s feeds 8 WMMAs per wave.
#define BM 128
#define BN 128
#define BK 32
#define LDT 40

template <int EPI>
__global__ __launch_bounds__(256) void k_gemm(const h16* __restrict__ A,
                                              const h16* __restrict__ W,
                                              const float* __restrict__ bias,
                                              void* __restrict__ Cout,
                                              int M, int N, int K) {
  __shared__ h16 sA[2][BM * LDT];
  __shared__ h16 sB[2][BN * LDT];
  int tid = threadIdx.x, lane = tid & 31, wave = tid >> 5;
  int wm = wave >> 2, wn = wave & 3;         // wave tile 64x32
  int bm = blockIdx.y * BM, bn = blockIdx.x * BN;
  v8f acc[4][2] = {};

  // this thread's slice of the 128x32-half tile: rows r0 and r0+64, 8 halves
  int r0 = tid >> 2;                 // 0..63
  int c0 = (tid & 3) * 8;            // 0,8,16,24
  const h16* gA = A + (size_t)(bm + r0) * K + c0;
  const h16* gB = W + (size_t)(bn + r0) * K + c0;
  int lo = r0 * LDT + c0, hi = (r0 + 64) * LDT + c0;
  int nst = K / BK;

  // prologue: stage 0 in flight
  async_copy_b128(gA,                  &sA[0][lo]);
  async_copy_b128(gA + (size_t)64 * K, &sA[0][hi]);
  async_copy_b128(gB,                  &sB[0][lo]);
  async_copy_b128(gB + (size_t)64 * K, &sB[0][hi]);

  for (int s = 0; s < nst; ++s) {
    int cur = s & 1;
    if (s + 1 < nst) {               // stream next stage while computing
      int nxt = cur ^ 1;
      const h16* pa = gA + (s + 1) * BK;
      const h16* pb = gB + (s + 1) * BK;
      async_copy_b128(pa,                  &sA[nxt][lo]);
      async_copy_b128(pa + (size_t)64 * K, &sA[nxt][hi]);
      async_copy_b128(pb,                  &sB[nxt][lo]);
      async_copy_b128(pb + (size_t)64 * K, &sB[nxt][hi]);
      wait_asynccnt<4>();            // stage s resident (in-order completion)
    } else {
      wait_asynccnt<0>();
    }
    __syncthreads();                 // all waves' stage-s data visible

    v16h bf0 = load_b_frag(sB[cur], LDT, wn * 32, 0);
    v16h bf1 = load_b_frag(sB[cur], LDT, wn * 32 + 16, 0);
#pragma unroll
    for (int sm = 0; sm < 4; ++sm) {
      v16h af = load_a_frag(sA[cur], LDT, wm * 64 + sm * 16, 0);
      acc[sm][0] = wmma16(af, bf0, acc[sm][0]);
      acc[sm][1] = wmma16(af, bf1, acc[sm][1]);
    }
    __syncthreads();                 // reads done; buffer may be refilled
  }

#pragma unroll
  for (int sm = 0; sm < 4; ++sm)
#pragma unroll
    for (int jn = 0; jn < 2; ++jn) {
      int n = bn + wn * 32 + jn * 16 + (lane & 15);
      float bv = bias[n];
      int mbase = bm + wm * 64 + sm * 16 + ((lane >> 4) << 3);
#pragma unroll
      for (int v = 0; v < 8; ++v) {
        float val = acc[sm][jn][v] + bv;
        if (EPI == 2) val = 0.5f * val * (1.f + erff(val * 0.70710678118f));
        size_t idx = (size_t)(mbase + v) * N + n;
        if (EPI == 1) ((float*)Cout)[idx] = val;
        else          ((h16*)Cout)[idx]  = (h16)val;
      }
    }
}

// --------------------------- fused window attention ------------------------
__device__ __forceinline__ int region8(int y) { return (y < 8) ? 0 : ((y < 12) ? 1 : 2); }

#define QLD 72
#define SLD 68

__global__ __launch_bounds__(128) void k_attn(const h16* __restrict__ qkv,
                                              const float* __restrict__ logit_scale,
                                              const float* __restrict__ tbl,
                                              h16* __restrict__ out) {
  __shared__ h16  sQ[64 * QLD];
  __shared__ h16  sK[64 * QLD];
  __shared__ h16  sVT[64 * QLD];
  __shared__ float sS[64 * SLD];
  __shared__ float sT[225];

  int tid = threadIdx.x, lane = tid & 31, wave = tid >> 5;
  int win = blockIdx.x >> 4, head = blockIdx.x & 15;
  const h16* qg = qkv + (size_t)win * 64 * 3072 + head * 64;

#pragma unroll
  for (int i = 0; i < 4; ++i) {              // stage q, k, v^T
    int lin = tid * 8 + i * 1024;
    int r = lin >> 6, c = lin & 63;
    *(v8h*)(sQ + r * QLD + c) = *(const v8h*)(qg + (size_t)r * 3072 + c);
    *(v8h*)(sK + r * QLD + c) = *(const v8h*)(qg + (size_t)r * 3072 + 1024 + c);
    v8h vv = *(const v8h*)(qg + (size_t)r * 3072 + 2048 + c);
#pragma unroll
    for (int e = 0; e < 8; ++e) sVT[(c + e) * QLD + r] = vv[e];
  }
  for (int i = tid; i < 225; i += 128) sT[i] = tbl[i * 16 + head];
  __syncthreads();

  // per-row L2 norm; clamped logit scale folded into q
  {
    float lsc = expf(fminf(logit_scale[head], 4.6051702f));   // ln(100)
    int r = tid & 63;
    h16* row = (tid < 64) ? (sQ + r * QLD) : (sK + r * QLD);
    float ss = 0.f;
    for (int d = 0; d < 64; ++d) { float q = (float)row[d]; ss += q * q; }
    float sc = 1.f / fmaxf(sqrtf(ss), 1e-12f);
    if (tid < 64) sc *= lsc;
    for (int d = 0; d < 64; ++d) row[d] = (h16)((float)row[d] * sc);
  }
  __syncthreads();

  int m0 = wave * 16;
  int wimg = win & 3, wy = wimg >> 1, wx = wimg & 1;
  v16h aq0 = load_a_frag(sQ, QLD, m0, 0);
  v16h aq1 = load_a_frag(sQ, QLD, m0, 32);
#pragma unroll
  for (int j = 0; j < 4; ++j) {              // S = Qn @ Kn^T (+bias +mask)
    v8f acc = {};
    acc = wmma16(aq0, load_b_frag(sK, QLD, j * 16, 0), acc);
    acc = wmma16(aq1, load_b_frag(sK, QLD, j * 16, 32), acc);
    int n = j * 16 + (lane & 15);
    int jy = n >> 3, jx = n & 7;
    int id_n = region8(wy * 8 + jy) * 3 + region8(wx * 8 + jx);
    int mloc = (lane >> 4) << 3;
#pragma unroll
    for (int v = 0; v < 8; ++v) {
      int m = m0 + mloc + v;
      int iy = m >> 3, ix = m & 7;
      int idx = (iy - jy + 7) * 15 + (ix - jx + 7);
      int id_m = region8(wy * 8 + iy) * 3 + region8(wx * 8 + ix);
      sS[m * SLD + n] = acc[v] + sT[idx] + ((id_m != id_n) ? -100.f : 0.f);
    }
  }
  __syncthreads();

  if (tid < 64) {                            // softmax row -> P (f16, reuse sQ)
    float mx = -1e30f;
    for (int n = 0; n < 64; ++n) mx = fmaxf(mx, sS[tid * SLD + n]);
    float sum = 0.f;
    for (int n = 0; n < 64; ++n) { float e = expf(sS[tid * SLD + n] - mx); sS[tid * SLD + n] = e; sum += e; }
    float inv = 1.f / sum;
    for (int n = 0; n < 64; ++n) sQ[tid * QLD + n] = (h16)(sS[tid * SLD + n] * inv);
  }
  __syncthreads();

  v16h ap0 = load_a_frag(sQ, QLD, m0, 0);
  v16h ap1 = load_a_frag(sQ, QLD, m0, 32);
#pragma unroll
  for (int j = 0; j < 4; ++j) {              // O = P @ V
    v8f acc = {};
    acc = wmma16(ap0, load_b_frag(sVT, QLD, j * 16, 0), acc);
    acc = wmma16(ap1, load_b_frag(sVT, QLD, j * 16, 32), acc);
    int n = j * 16 + (lane & 15);
    int mbase = m0 + ((lane >> 4) << 3);
#pragma unroll
    for (int v = 0; v < 8; ++v)
      out[(size_t)(win * 64 + mbase + v) * DIMC + head * 64 + n] = (h16)acc[v];
  }
}

// ------------------- LayerNorm + residual epilogues ------------------------
__device__ __forceinline__ float block_reduce_sum(float v, float* red) {
  int tid = threadIdx.x;
  red[tid] = v;
  __syncthreads();
  for (int s = 128; s > 0; s >>= 1) { if (tid < s) red[tid] += red[tid + s]; __syncthreads(); }
  float r = red[0];
  __syncthreads();
  return r;
}

// window-reverse + roll(+4,+4) + LN + residual; writes xs (f32 -> d_out, f16 -> ws)
__global__ __launch_bounds__(256) void k_res_ln1(const float* __restrict__ x,
                                                 const float* __restrict__ proj,
                                                 const float* __restrict__ g,
                                                 const float* __restrict__ b,
                                                 float* __restrict__ of32,
                                                 h16* __restrict__ of16) {
  __shared__ float red[256];
  int gid = blockIdx.x;
  int bimg = gid >> 8, pos = gid & 255;
  int y = pos >> 4, xc = pos & 15;
  int yp = (y + 12) & 15, xp = (xc + 12) & 15;
  int src = (bimg * 4 + (yp >> 3) * 2 + (xp >> 3)) * 64 + (yp & 7) * 8 + (xp & 7);
  const float* prow = proj + (size_t)src * DIMC;
  const float* xrow = x + (size_t)gid * DIMC;

  float s = 0.f;
  for (int c = threadIdx.x; c < DIMC; c += 256) s += prow[c];
  float mean = block_reduce_sum(s, red) * (1.f / DIMC);
  float vs = 0.f;
  for (int c = threadIdx.x; c < DIMC; c += 256) { float d = prow[c] - mean; vs += d * d; }
  float rstd = rsqrtf(block_reduce_sum(vs, red) * (1.f / DIMC) + 1e-5f);
  for (int c = threadIdx.x; c < DIMC; c += 256) {
    float o = xrow[c] + ((prow[c] - mean) * rstd * g[c] + b[c]);
    of32[(size_t)gid * DIMC + c] = o;
    of16[(size_t)gid * DIMC + c] = (h16)o;
  }
}

// xs += LN(fc2out); in-place on d_out (each block owns its row)
__global__ __launch_bounds__(256) void k_res_ln2(float* __restrict__ xs,
                                                 const float* __restrict__ h,
                                                 const float* __restrict__ g,
                                                 const float* __restrict__ b) {
  __shared__ float red[256];
  int gid = blockIdx.x;
  const float* hrow = h + (size_t)gid * DIMC;
  float* xrow = xs + (size_t)gid * DIMC;
  float s = 0.f;
  for (int c = threadIdx.x; c < DIMC; c += 256) s += hrow[c];
  float mean = block_reduce_sum(s, red) * (1.f / DIMC);
  float vs = 0.f;
  for (int c = threadIdx.x; c < DIMC; c += 256) { float d = hrow[c] - mean; vs += d * d; }
  float rstd = rsqrtf(block_reduce_sum(vs, red) * (1.f / DIMC) + 1e-5f);
  for (int c = threadIdx.x; c < DIMC; c += 256)
    xrow[c] = xrow[c] + ((hrow[c] - mean) * rstd * g[c] + b[c]);
}

// ------------------------------ launcher -----------------------------------
extern "C" void kernel_launch(void* const* d_in, const int* in_sizes, int n_in,
                              void* d_out, int out_size, void* d_ws, size_t ws_size,
                              hipStream_t stream) {
  const float* x      = (const float*)d_in[0];
  const float* qkv_w  = (const float*)d_in[1];
  const float* q_bias = (const float*)d_in[2];
  const float* v_bias = (const float*)d_in[3];
  const float* lscale = (const float*)d_in[4];
  const float* cpb_w1 = (const float*)d_in[5];
  const float* cpb_b1 = (const float*)d_in[6];
  const float* cpb_w2 = (const float*)d_in[7];
  const float* proj_w = (const float*)d_in[8];
  const float* proj_b = (const float*)d_in[9];
  const float* n1w    = (const float*)d_in[10];
  const float* n1b    = (const float*)d_in[11];
  const float* fc1_w  = (const float*)d_in[12];
  const float* fc1_b  = (const float*)d_in[13];
  const float* fc2_w  = (const float*)d_in[14];
  const float* fc2_b  = (const float*)d_in[15];
  const float* n2w    = (const float*)d_in[16];
  const float* n2b    = (const float*)d_in[17];

  // workspace layout (lifetime-aliased):
  //   XW   (xw f16 for QKV A; dead after QKV) ==reused==> xs f16 for FC1 A
  //   QKV  (qkv f16; dead after attention)    ==reused==> h f16 (FC1 out)
  //   PROJ (proj f32; dead after ln1)         ==reused==> fc2 out f32
  char* ws = (char*)d_ws;
  const size_t OFF_XW   = 0;                                  // 64 MiB
  const size_t OFF_WQKV = OFF_XW   + (size_t)TOK * DIMC * 2;  // 6 MiB
  const size_t OFF_WPRJ = OFF_WQKV + (size_t)3072 * DIMC * 2; // 2 MiB
  const size_t OFF_WFC1 = OFF_WPRJ + (size_t)DIMC * DIMC * 2; // 8 MiB
  const size_t OFF_WFC2 = OFF_WFC1 + (size_t)HIDDEN_ * DIMC * 2;
  const size_t OFF_B3   = OFF_WFC2 + (size_t)DIMC * HIDDEN_ * 2;
  const size_t OFF_TBL  = OFF_B3   + 16384;
  const size_t OFF_QKV  = OFF_TBL  + 16384;                   // 256 MiB (qkv/h)
  const size_t OFF_ATT  = OFF_QKV  + (size_t)TOK * HIDDEN_ * 2;
  const size_t OFF_PRJ  = OFF_ATT  + (size_t)TOK * DIMC * 2;  // 128 MiB (proj/fc2)

  h16*   xw    = (h16*)(ws + OFF_XW);
  h16*   xs16  = (h16*)(ws + OFF_XW);     // alias after QKV consumed
  h16*   wqkv  = (h16*)(ws + OFF_WQKV);
  h16*   wprj  = (h16*)(ws + OFF_WPRJ);
  h16*   wfc1  = (h16*)(ws + OFF_WFC1);
  h16*   wfc2  = (h16*)(ws + OFF_WFC2);
  float* b3    = (float*)(ws + OFF_B3);
  float* tbl   = (float*)(ws + OFF_TBL);
  h16*   qkv16 = (h16*)(ws + OFF_QKV);
  h16*   h16b  = (h16*)(ws + OFF_QKV);    // alias after attention consumed
  h16*   att16 = (h16*)(ws + OFF_ATT);
  float* prjf  = (float*)(ws + OFF_PRJ);
  float* fc2f  = (float*)(ws + OFF_PRJ);  // alias after ln1 consumed
  float* out   = (float*)d_out;

  // 1) weights -> f16, qkv bias vector, CPB table, shifted-window gather
  k_f32_to_f16<<<3072, 256, 0, stream>>>(qkv_w, wqkv, 3072 * DIMC / 4);
  k_f32_to_f16<<<1024, 256, 0, stream>>>(proj_w, wprj, DIMC * DIMC / 4);
  k_f32_to_f16<<<4096, 256, 0, stream>>>(fc1_w, wfc1, HIDDEN_ * DIMC / 4);
  k_f32_to_f16<<<4096, 256, 0, stream>>>(fc2_w, wfc2, DIMC * HIDDEN_ / 4);
  k_build_qkv_bias<<<12, 256, 0, stream>>>(q_bias, v_bias, b3);
  k_cpb<<<225, 128, 0, stream>>>(cpb_w1, cpb_b1, cpb_w2, tbl);
  k_gather_x<<<TOK, 256, 0, stream>>>(x, xw);

  // 2) QKV GEMM: [32768,1024] x [3072,1024]^T -> f16
  k_gemm<0><<<dim3(3072 / BN, TOK / BM), 256, 0, stream>>>(xw, wqkv, b3, qkv16, TOK, 3072, DIMC);

  // 3) fused cosine attention per (window, head)
  k_attn<<<NWIN * HEADS_, 128, 0, stream>>>(qkv16, lscale, tbl, att16);

  // 4) proj GEMM -> f32 (still window order)
  k_gemm<1><<<dim3(DIMC / BN, TOK / BM), 256, 0, stream>>>(att16, wprj, proj_b, prjf, TOK, DIMC, DIMC);

  // 5) window-reverse + roll + LN + residual -> d_out (f32) and xs f16
  k_res_ln1<<<TOK, 256, 0, stream>>>(x, prjf, n1w, n1b, out, xs16);

  // 6) FC1 (GELU fused) -> f16, FC2 -> f32
  k_gemm<2><<<dim3(HIDDEN_ / BN, TOK / BM), 256, 0, stream>>>(xs16, wfc1, fc1_b, h16b, TOK, HIDDEN_, DIMC);
  k_gemm<1><<<dim3(DIMC / BN, TOK / BM), 256, 0, stream>>>(h16b, wfc2, fc2_b, fc2f, TOK, DIMC, HIDDEN_);

  // 7) final LN + residual in-place on d_out
  k_res_ln2<<<TOK, 256, 0, stream>>>(out, fc2f, n2w, n2b);
}
